// LorentzianTransformerLayer_12421045420096
// MI455X (gfx1250) — matrix operand order repaired and checked by reading, not compiled, not run on hardware
//
#include <hip/hip_runtime.h>

typedef unsigned short u16;
typedef unsigned int u32;
typedef __attribute__((ext_vector_type(16))) __bf16 bf16x16;
typedef __attribute__((ext_vector_type(8)))  float  v8f;
typedef __attribute__((ext_vector_type(4)))  u32    u32x4;
typedef __attribute__((ext_vector_type(8)))  int    i32x8;
typedef __attribute__((ext_vector_type(4)))  int    i32x4;

#if defined(__has_builtin)
#if __has_builtin(__builtin_amdgcn_tensor_load_to_lds) && __has_builtin(__builtin_amdgcn_s_wait_tensorcnt)
#define USE_TDM 1
#endif
#endif
#ifndef USE_TDM
#define USE_TDM 0
#endif

union AF {
    bf16x16 v;
    u32  u[8];
    uint2 u2[4];
    u16  s[16];
};

__device__ __forceinline__ u16 f2bf(float f) {
    u32 x = __float_as_uint(f);
    u32 r = x + 0x7FFFu + ((x >> 16) & 1u);
    return (u16)(r >> 16);
}

__device__ __forceinline__ v8f vzero8() {
    v8f z;
#pragma unroll
    for (int i = 0; i < 8; ++i) z[i] = 0.0f;
    return z;
}

// K index base for A-fragment VGPR v (contiguous order: k = 16*half + 2*v)
__device__ __forceinline__ int a_kb(int v, int half) {
    return 2 * v + half * 16;
}

static constexpr int BB  = 4;
static constexpr int TT  = 2048;
static constexpr int DD  = 1024;
static constexpr int HH  = 16;
static constexpr int DKK = 64;
static constexpr int DFF = 4096;
static constexpr int MM  = BB * TT; // 8192

// ---------------------------------------------------------------------------
// elementwise f32 -> bf16
// ---------------------------------------------------------------------------
__global__ void cvt_bf16_kernel(const float* __restrict__ in, u16* __restrict__ out, long n) {
    long i = (long)blockIdx.x * blockDim.x + threadIdx.x;
    long stride = (long)gridDim.x * blockDim.x;
    for (; i < n; i += stride) out[i] = f2bf(in[i]);
}

// ---------------------------------------------------------------------------
// transpose + convert: in[K][N] f32 -> out[N][K] bf16   (K,N multiples of 32)
// ---------------------------------------------------------------------------
__global__ void transpose_cvt_kernel(const float* __restrict__ in, u16* __restrict__ out,
                                     int K, int N) {
    __shared__ float tile[32][33];
    int tx = threadIdx.x & 31;
    int ty = threadIdx.x >> 5; // 0..7
    int n0 = blockIdx.x * 32;
    int k0 = blockIdx.y * 32;
#pragma unroll
    for (int r = ty; r < 32; r += 8)
        tile[r][tx] = in[(size_t)(k0 + r) * N + (n0 + tx)];
    __syncthreads();
#pragma unroll
    for (int r = ty; r < 32; r += 8)
        out[(size_t)(n0 + r) * K + (k0 + tx)] = f2bf(tile[tx][r]);
}

// ---------------------------------------------------------------------------
// tau / |x|^2 extraction from mu
// ---------------------------------------------------------------------------
__global__ void taux_kernel(const float* __restrict__ mu, float* __restrict__ tau,
                            float* __restrict__ xsq) {
    int i = blockIdx.x * blockDim.x + threadIdx.x;
    if (i >= MM) return;
    const float* m = mu + (size_t)i * 129;
    tau[i] = m[0];
    float s = 0.f;
#pragma unroll 8
    for (int d = 0; d < 128; ++d) { float v = m[1 + d]; s += v * v; }
    xsq[i] = s;
}

// ---------------------------------------------------------------------------
// packed Lorentzian causal mask: bit(b,i,j) = (ds2<0 && tau_i-tau_j<=0)
// one block per (b,i) row; 256 threads sweep j; ballot packs 32 bits/word
// ---------------------------------------------------------------------------
__global__ void mask_kernel(const float* __restrict__ mu, const float* __restrict__ tau,
                            const float* __restrict__ xsq, u32* __restrict__ mask) {
    __shared__ float xi[128];
    int bi = blockIdx.x;          // b*T + i
    int b  = bi / TT;
    const float* mrow = mu + (size_t)bi * 129;
    if (threadIdx.x < 128) xi[threadIdx.x] = mrow[1 + threadIdx.x];
    __syncthreads();
    float ti = tau[bi];
    float xsqi = xsq[bi];
    int lane = threadIdx.x & 31;
    for (int it = 0; it < TT / 256; ++it) {
        int j = it * 256 + threadIdx.x;
        int bj = b * TT + j;
        const float* mj = mu + (size_t)bj * 129 + 1;
        float dot = 0.f;
#pragma unroll 8
        for (int d = 0; d < 128; ++d) dot += xi[d] * mj[d];
        float td = ti - tau[bj];
        float dist = xsqi + xsq[bj] - 2.f * dot;
        bool pred = ((dist - td * td) < 0.f) && (td <= 0.f);
        u32 w = __builtin_amdgcn_ballot_w32(pred);
        if (lane == 0) mask[(size_t)bi * (TT / 32) + (j >> 5)] = w;
    }
}

// ---------------------------------------------------------------------------
// Generic bf16 WMMA GEMM:  C[M,N] = A[M,K] @ Bt[N,K]^T + bias
// mode: 0 -> f32 out, 1 -> bf16 out, 2 -> gelu then bf16 out
// Block tile 128x128, 8 waves (2x4), wave tile 64x32, K-step 32.
// Double-buffered LDS; staging via Tensor Data Mover (TDM) when available:
// wave0 issues tensor_load_to_lds for the NEXT k-slice while all waves run
// WMMA on the current one; one barrier per k-step.
// ---------------------------------------------------------------------------
__global__ void gemm_bf16_kernel(const u16* __restrict__ A, const u16* __restrict__ Bt,
                                 const float* __restrict__ bias,
                                 float* __restrict__ Cf, u16* __restrict__ Cb,
                                 int M, int N, int K, int mode) {
    constexpr int LDA = 40; // 32 elems + 8 pad (row = 64B data + 16B pad)
    __shared__ u16 As[2][128 * LDA];
    __shared__ u16 Bs[2][128 * LDA];

    int tid  = threadIdx.x;
    int w    = tid >> 5;
    int lane = tid & 31;
    int wm   = w >> 2;    // 0..1
    int wn   = w & 3;     // 0..3
    int lm   = lane & 15;
    int half = lane >> 4;

    int blockRow = blockIdx.y * 128;
    int blockCol = blockIdx.x * 128;

    v8f acc[4][2];
#pragma unroll
    for (int i = 0; i < 4; ++i)
#pragma unroll
        for (int j = 0; j < 2; ++j) acc[i][j] = vzero8();

#if USE_TDM
    // TDM descriptor: 2-D tensor, 2-byte elements, tile 32(K) x 128(rows),
    // row stride K elements; LDS padding: every 16 DWORDs (64B) insert 4
    // DWORDs (16B)  ==>  reproduces the LDA=40 padded layout.
    auto issueTile = [&](const u16* gbase, int row0, int kk, u16* lds) {
        unsigned long long ga = (unsigned long long)(uintptr_t)(gbase + (size_t)row0 * K + kk);
        u32 ldsOff = (u32)(uintptr_t)lds; // low 32 bits of generic addr = LDS offset
        u32x4 g0;
        g0[0] = 1u;                                   // count=1 (valid descriptor)
        g0[1] = ldsOff;                               // lds_addr
        g0[2] = (u32)ga;                              // global_addr[31:0]
        g0[3] = ((u32)(ga >> 32) & 0x1FFFFFFu) | (2u << 30); // [56:32] | type=2
        i32x8 g1;
        g1[0] = (int)((1u << 16)      // data_size = 2 bytes
                    | (1u << 20)      // pad_enable
                    | (3u << 22)      // pad_interval: 16 DWORDs
                    | (3u << 25));    // pad_amount: 4 DWORDs
        g1[1] = (int)(((u32)K & 0xFFFFu) << 16);          // tensor_dim0[15:0]
        g1[2] = (int)(((u32)K >> 16) & 0xFFFFu);          // tensor_dim0[31:16], tensor_dim1 lo16=0
        g1[3] = (int)(16u | (32u << 16));                 // tensor_dim1 = 1<<20 ; tile_dim0 = 32
        g1[4] = (int)128u;                                // tile_dim1 = 128 rows
        g1[5] = (int)(u32)K;                              // tensor_dim0_stride lo32
        g1[6] = 0;                                        // stride hi / dim1_stride lo
        g1[7] = 0;
        i32x4 gz4 = {0, 0, 0, 0};
        i32x8 gz8 = {0, 0, 0, 0, 0, 0, 0, 0};
        __builtin_amdgcn_tensor_load_to_lds(g0, g1, gz4, gz4, gz8, 0);
    };
    if (tid == 0) {
        issueTile(A, blockRow, 0, As[0]);
        issueTile(Bt, blockCol, 0, Bs[0]);
    }
#endif

    int srow = tid >> 1;
    int sseg = (tid & 1) * 16;
    (void)srow; (void)sseg;

    for (int k0 = 0; k0 < K; k0 += 32) {
        int cur;
#if USE_TDM
        cur = (k0 >> 5) & 1;
        if (w == 0) __builtin_amdgcn_s_wait_tensorcnt(0);
        __syncthreads();
        if (tid == 0 && (k0 + 32) < K) {
            issueTile(A, blockRow, k0 + 32, As[cur ^ 1]);
            issueTile(Bt, blockCol, k0 + 32, Bs[cur ^ 1]);
        }
#else
        cur = 0;
        __syncthreads();
        {
            const uint2* ga = (const uint2*)(A + (size_t)(blockRow + srow) * K + k0 + sseg);
            uint2* la = (uint2*)&As[0][srow * LDA + sseg];
#pragma unroll
            for (int q = 0; q < 4; ++q) la[q] = ga[q];
            const uint2* gb = (const uint2*)(Bt + (size_t)(blockCol + srow) * K + k0 + sseg);
            uint2* lb = (uint2*)&Bs[0][srow * LDA + sseg];
#pragma unroll
            for (int q = 0; q < 4; ++q) lb[q] = gb[q];
        }
        __syncthreads();
#endif

        AF af[4];
#pragma unroll
        for (int tm = 0; tm < 4; ++tm) {
            int row = wm * 64 + tm * 16 + lm;
            int base = row * LDA + half * 16;
            af[tm].u2[0] = *(const uint2*)&As[cur][base];
            af[tm].u2[1] = *(const uint2*)&As[cur][base + 4];
            af[tm].u2[2] = *(const uint2*)&As[cur][base + 8];
            af[tm].u2[3] = *(const uint2*)&As[cur][base + 12];
        }
        AF bf[2];
#pragma unroll
        for (int tn = 0; tn < 2; ++tn) {
            int row = wn * 32 + tn * 16 + lm;
            int base = row * LDA + half * 16;
            bf[tn].u2[0] = *(const uint2*)&Bs[cur][base];
            bf[tn].u2[1] = *(const uint2*)&Bs[cur][base + 4];
            bf[tn].u2[2] = *(const uint2*)&Bs[cur][base + 8];
            bf[tn].u2[3] = *(const uint2*)&Bs[cur][base + 12];
        }
#pragma unroll
        for (int tm = 0; tm < 4; ++tm)
#pragma unroll
            for (int tn = 0; tn < 2; ++tn)
                acc[tm][tn] = __builtin_amdgcn_wmma_f32_16x16x32_bf16(
                    false, af[tm].v, false, bf[tn].v, (short)0, acc[tm][tn], false, false);
    }

    // epilogue
#pragma unroll
    for (int tm = 0; tm < 4; ++tm) {
#pragma unroll
        for (int tn = 0; tn < 2; ++tn) {
#pragma unroll
            for (int r = 0; r < 8; ++r) {
                int m = blockRow + wm * 64 + tm * 16 + r + half * 8;
                int n = blockCol + wn * 32 + tn * 16 + lm;
                float val = acc[tm][tn][r] + bias[n];
                if (mode == 2)
                    val = 0.5f * val * (1.0f + erff(val * 0.70710678118654752f));
                if (mode == 0) Cf[(size_t)m * N + n] = val;
                else           Cb[(size_t)m * N + n] = f2bf(val);
            }
        }
    }
}

// ---------------------------------------------------------------------------
// Flash attention with Lorentzian mask.
// grid: (T/128, H, B); 8 waves/block, each wave owns 16 query rows.
// Q,K,V bf16 [B*T][D] (head h at col h*64). Output bf16 [B*T][D].
// ---------------------------------------------------------------------------
__global__ void attn_kernel(const u16* __restrict__ Qb, const u16* __restrict__ Kb,
                            const u16* __restrict__ Vb, const u32* __restrict__ mask,
                            u16* __restrict__ Ob) {
    constexpr int LDK = 72; // 64 + 8
    constexpr int LDV = 40; // 32 + 8
    __shared__ u16 KsS[32 * LDK];    // K tile: [key 32][dk 64]
    __shared__ u16 VtS[64 * LDV];    // V^T tile: [dk 64][key 32]
    __shared__ u16 PsS[8 * 16 * 32]; // per-wave P scratch [m 16][k 32]

    int tid  = threadIdx.x;
    int w    = tid >> 5;
    int lane = tid & 31;
    int lm   = lane & 15;
    int half = lane >> 4;

    int b    = blockIdx.z;
    int hd0  = blockIdx.y * DKK;
    int q0   = blockIdx.x * 128 + w * 16;

    u16* ps = &PsS[w * 512];

    // load Q fragments (16 rows x 64 K), prefetched once
    AF qf[2];
#pragma unroll
    for (int s = 0; s < 2; ++s) {
        size_t rowoff = (size_t)(b * TT + q0 + lm) * DD + hd0 + s * 32;
#pragma unroll
        for (int v = 0; v < 8; ++v)
            qf[s].u[v] = *(const u32*)(Qb + rowoff + a_kb(v, half));
    }

    v8f o[4];
#pragma unroll
    for (int d = 0; d < 4; ++d) o[d] = vzero8();
    float rowM[8], rowL[8];
#pragma unroll
    for (int r = 0; r < 8; ++r) { rowM[r] = -INFINITY; rowL[r] = 0.f; }

    const float scale = 0.125f; // 1/sqrt(64)

    for (int j0 = 0; j0 < TT; j0 += 32) {
        __syncthreads();
        // stage K tile (row-major) and V tile (transposed)
        {
            int krow = tid >> 3;
            int kseg = (tid & 7) * 8;
            const uint2* g = (const uint2*)(Kb + (size_t)(b * TT + j0 + krow) * DD + hd0 + kseg);
            uint2* l = (uint2*)&KsS[krow * LDK + kseg];
            l[0] = g[0]; l[1] = g[1];

            int key = tid & 31;
            int dk0 = (tid >> 5) * 8;
            union { uint2 a[2]; u16 s[8]; } tv;
            const uint2* gv = (const uint2*)(Vb + (size_t)(b * TT + j0 + key) * DD + hd0 + dk0);
            tv.a[0] = gv[0]; tv.a[1] = gv[1];
#pragma unroll
            for (int q = 0; q < 8; ++q) VtS[(dk0 + q) * LDV + key] = tv.s[q];
        }
        __syncthreads();

        // S = Q @ K^T  (two 16x16 key tiles)
        v8f sacc[2];
#pragma unroll
        for (int tn = 0; tn < 2; ++tn) {
            sacc[tn] = vzero8();
#pragma unroll
            for (int s = 0; s < 2; ++s) {
                AF kf;
                int base = (tn * 16 + lm) * LDK + s * 32 + half * 16;
                kf.u2[0] = *(const uint2*)&KsS[base];
                kf.u2[1] = *(const uint2*)&KsS[base + 4];
                kf.u2[2] = *(const uint2*)&KsS[base + 8];
                kf.u2[3] = *(const uint2*)&KsS[base + 12];
                sacc[tn] = __builtin_amdgcn_wmma_f32_16x16x32_bf16(
                    false, qf[s].v, false, kf.v, (short)0, sacc[tn], false, false);
            }
        }

        // masked online softmax over the 32-key slab
#pragma unroll
        for (int r = 0; r < 8; ++r) {
            int i_row = q0 + r + half * 8;
            u32 wm_ = mask[((size_t)(b * TT + i_row) << 6) | (j0 >> 5)];
            int v0 = (wm_ >> lm) & 1;
            int v1 = (wm_ >> (16 + lm)) & 1;
            float s0 = sacc[0][r] * scale;
            float s1 = sacc[1][r] * scale;
            float m0 = v0 ? s0 : -INFINITY;
            float m1 = v1 ? s1 : -INFINITY;
            float tmax = fmaxf(m0, m1);
#pragma unroll
            for (int off = 1; off < 16; off <<= 1)
                tmax = fmaxf(tmax, __shfl_xor(tmax, off, 16));
            float mNew = fmaxf(rowM[r], tmax);
            float alpha = (mNew == -INFINITY) ? 1.0f : __expf(rowM[r] - mNew);
            float p0 = v0 ? __expf(s0 - mNew) : 0.f;
            float p1 = v1 ? __expf(s1 - mNew) : 0.f;
            float psum = p0 + p1;
#pragma unroll
            for (int off = 1; off < 16; off <<= 1)
                psum += __shfl_xor(psum, off, 16);
            rowL[r] = rowL[r] * alpha + psum;
            rowM[r] = mNew;
#pragma unroll
            for (int d = 0; d < 4; ++d) o[d][r] *= alpha;
            int m = r + half * 8;
            ps[m * 32 + lm]      = f2bf(p0);
            ps[m * 32 + 16 + lm] = f2bf(p1);
        }

        // reload P in A-fragment layout (same-wave LDS ops stay in order)
        AF pf;
#pragma unroll
        for (int v = 0; v < 8; ++v)
            pf.u[v] = *(const u32*)&ps[lm * 32 + a_kb(v, half)];

        // O += P @ V  (4 dk tiles of 16)
#pragma unroll
        for (int d = 0; d < 4; ++d) {
            AF vf;
            int base = (d * 16 + lm) * LDV + half * 16;
            vf.u2[0] = *(const uint2*)&VtS[base];
            vf.u2[1] = *(const uint2*)&VtS[base + 4];
            vf.u2[2] = *(const uint2*)&VtS[base + 8];
            vf.u2[3] = *(const uint2*)&VtS[base + 12];
            o[d] = __builtin_amdgcn_wmma_f32_16x16x32_bf16(
                false, pf.v, false, vf.v, (short)0, o[d], false, false);
        }
    }

    // finalize: divide by row sums (zero rows with no causal key) and store bf16
#pragma unroll
    for (int r = 0; r < 8; ++r) {
        float inv = (rowL[r] == 0.f) ? 0.f : 1.0f / rowL[r];
        int i_row = q0 + r + half * 8;
        size_t base = (size_t)(b * TT + i_row) * DD + hd0 + lm;
#pragma unroll
        for (int d = 0; d < 4; ++d)
            Ob[base + d * 16] = f2bf(o[d][r] * inv);
    }
}

// ---------------------------------------------------------------------------
// out = LayerNorm(x + y) * g + be ; optional bf16 copy
// one block per row of 1024
// ---------------------------------------------------------------------------
__global__ void add_ln_kernel(const float* __restrict__ x, const float* __restrict__ y,
                              const float* __restrict__ g, const float* __restrict__ be,
                              float* __restrict__ outF, u16* __restrict__ outB) {
    __shared__ float red[256];
    int tid = threadIdx.x;
    size_t base = (size_t)blockIdx.x * DD;
    float v[4];
    float s = 0.f;
#pragma unroll
    for (int q = 0; q < 4; ++q) {
        int c = tid + q * 256;
        v[q] = x[base + c] + y[base + c];
        s += v[q];
    }
    red[tid] = s; __syncthreads();
    for (int st = 128; st > 0; st >>= 1) { if (tid < st) red[tid] += red[tid + st]; __syncthreads(); }
    float mean = red[0] * (1.0f / DD);
    __syncthreads();
    float s2 = 0.f;
#pragma unroll
    for (int q = 0; q < 4; ++q) { float d = v[q] - mean; s2 += d * d; }
    red[tid] = s2; __syncthreads();
    for (int st = 128; st > 0; st >>= 1) { if (tid < st) red[tid] += red[tid + st]; __syncthreads(); }
    float var = red[0] * (1.0f / DD);
    float rst = rsqrtf(var + 1e-5f);
#pragma unroll
    for (int q = 0; q < 4; ++q) {
        int c = tid + q * 256;
        float o = (v[q] - mean) * rst * g[c] + be[c];
        outF[base + c] = o;
        if (outB) outB[base + c] = f2bf(o);
    }
}

// ---------------------------------------------------------------------------
extern "C" void kernel_launch(void* const* d_in, const int* in_sizes, int n_in,
                              void* d_out, int out_size, void* d_ws, size_t ws_size,
                              hipStream_t stream) {
    (void)in_sizes; (void)n_in; (void)out_size; (void)ws_size;
    const float* h   = (const float*)d_in[0];
    const float* mu  = (const float*)d_in[1];
    const float* Wq  = (const float*)d_in[2];
    const float* bq  = (const float*)d_in[3];
    const float* Wk  = (const float*)d_in[4];
    const float* bk  = (const float*)d_in[5];
    const float* Wv  = (const float*)d_in[6];
    const float* bv  = (const float*)d_in[7];
    const float* Wo  = (const float*)d_in[8];
    const float* bo  = (const float*)d_in[9];
    const float* W1  = (const float*)d_in[10];
    const float* b1  = (const float*)d_in[11];
    const float* W2  = (const float*)d_in[12];
    const float* b2  = (const float*)d_in[13];
    const float* g1  = (const float*)d_in[14];
    const float* be1 = (const float*)d_in[15];
    const float* g2  = (const float*)d_in[16];
    const float* be2 = (const float*)d_in[17];
    float* out = (float*)d_out;

    unsigned char* ws = (unsigned char*)d_ws;
    size_t off = 0;
    auto take = [&](size_t bytes) -> void* {
        void* p = ws + off;
        off = (off + bytes + 255) & ~(size_t)255;
        return p;
    };
    u16* hB    = (u16*)take((size_t)MM * DD * 2);
    u16* WqT   = (u16*)take((size_t)DD * DD * 2);
    u16* WkT   = (u16*)take((size_t)DD * DD * 2);
    u16* WvT   = (u16*)take((size_t)DD * DD * 2);
    u16* WoT   = (u16*)take((size_t)DD * DD * 2);
    u16* W1T   = (u16*)take((size_t)DFF * DD * 2);
    u16* W2T   = (u16*)take((size_t)DD * DFF * 2);
    u16* Qb    = (u16*)take((size_t)MM * DD * 2);
    u16* Kb    = (u16*)take((size_t)MM * DD * 2);
    u16* Vb    = (u16*)take((size_t)MM * DD * 2);
    u16* Ob    = (u16*)take((size_t)MM * DD * 2);
    float* hattn = (float*)take((size_t)MM * DD * 4);
    float* h1    = (float*)take((size_t)MM * DD * 4);
    u16* h1b   = (u16*)take((size_t)MM * DD * 2);
    u16* ffn1  = (u16*)take((size_t)MM * DFF * 2);
    float* hffn  = (float*)take((size_t)MM * DD * 4);
    float* tau   = (float*)take((size_t)MM * 4);
    float* xsq   = (float*)take((size_t)MM * 4);
    u32* mask  = (u32*)take((size_t)MM * (TT / 32) * 4);

    // 1. precision conversion / weight transposition
    cvt_bf16_kernel<<<8192, 256, 0, stream>>>(h, hB, (long)MM * DD);
    transpose_cvt_kernel<<<dim3(DD / 32, DD / 32), 256, 0, stream>>>(Wq, WqT, DD, DD);
    transpose_cvt_kernel<<<dim3(DD / 32, DD / 32), 256, 0, stream>>>(Wk, WkT, DD, DD);
    transpose_cvt_kernel<<<dim3(DD / 32, DD / 32), 256, 0, stream>>>(Wv, WvT, DD, DD);
    transpose_cvt_kernel<<<dim3(DD / 32, DD / 32), 256, 0, stream>>>(Wo, WoT, DD, DD);
    transpose_cvt_kernel<<<dim3(DFF / 32, DD / 32), 256, 0, stream>>>(W1, W1T, DD, DFF);
    transpose_cvt_kernel<<<dim3(DD / 32, DFF / 32), 256, 0, stream>>>(W2, W2T, DFF, DD);

    // 2. Lorentzian causal mask (packed bits)
    taux_kernel<<<MM / 256, 256, 0, stream>>>(mu, tau, xsq);
    mask_kernel<<<MM, 256, 0, stream>>>(mu, tau, xsq, mask);

    // 3. QKV projections (bf16 out)
    gemm_bf16_kernel<<<dim3(DD / 128, MM / 128), 256, 0, stream>>>(hB, WqT, bq, nullptr, Qb, MM, DD, DD, 1);
    gemm_bf16_kernel<<<dim3(DD / 128, MM / 128), 256, 0, stream>>>(hB, WkT, bk, nullptr, Kb, MM, DD, DD, 1);
    gemm_bf16_kernel<<<dim3(DD / 128, MM / 128), 256, 0, stream>>>(hB, WvT, bv, nullptr, Vb, MM, DD, DD, 1);

    // 4. flash attention with Lorentzian mask
    attn_kernel<<<dim3(TT / 128, HH, BB), 256, 0, stream>>>(Qb, Kb, Vb, mask, Ob);

    // 5. output projection (f32 out)
    gemm_bf16_kernel<<<dim3(DD / 128, MM / 128), 256, 0, stream>>>(Ob, WoT, bo, hattn, nullptr, MM, DD, DD, 0);

    // 6. h1 = LN(h + h_attn)
    add_ln_kernel<<<MM, 256, 0, stream>>>(h, hattn, g1, be1, h1, h1b);

    // 7. FFN
    gemm_bf16_kernel<<<dim3(DFF / 128, MM / 128), 256, 0, stream>>>(h1b, W1T, b1, nullptr, ffn1, MM, DFF, DD, 2);
    gemm_bf16_kernel<<<dim3(DD / 128, MM / 128), 256, 0, stream>>>(ffn1, W2T, b2, hffn, nullptr, MM, DD, DFF, 0);

    // 8. out = LN(h1 + h_ffn)
    add_ln_kernel<<<MM, 256, 0, stream>>>(h1, hffn, g2, be2, out, nullptr);
}
